// Attention_35210141892943
// MI455X (gfx1250) — matrix operand order repaired and checked by reading, not compiled, not run on hardware
//
#include <hip/hip_runtime.h>

#define DEV __device__ __forceinline__

typedef __attribute__((ext_vector_type(16))) _Float16 v16h;
typedef __attribute__((ext_vector_type(8)))  _Float16 v8h;
typedef __attribute__((ext_vector_type(8)))  float    v8f;
typedef __attribute__((ext_vector_type(4)))  float    v4f;

constexpr int B_  = 2;
constexpr int T_  = 4096;
constexpr int C_  = 128;
constexpr int H_  = 4;
constexpr int DH_ = 32;
constexpr int M_  = B_ * T_;     // 8192 token rows
constexpr int N3_ = 3 * C_;      // 384 qkv outputs
// 1/sqrt(32) * log2(e): scores computed in log2 domain so softmax uses exp2
constexpr float QSCALE_ = 0.17677669529663687f * 1.4426950408889634f;

union Frag16 { v16h v; v8h h8[2]; _Float16 h[16]; };

DEV v8f wmma_f16(v16h a, v16h b, v8f c) {
  // D = A(16x32 f16) * B(32x16 f16) + C(16x16 f32)
  return __builtin_amdgcn_wmma_f32_16x16x32_f16(
      /*neg_a=*/false, a, /*neg_b=*/false, b,
      /*c_mod=*/(short)0, c, /*reuse_a=*/false, /*reuse_b=*/false);
}

// A-fragment from f32 memory: halves 0..7 <- p[0..7], halves 8..15 <- p[16..23]
DEV v16h cvt_a_f32(const float* p) {
  v4f a0 = *(const v4f*)(p),      a1 = *(const v4f*)(p + 4);
  v4f a2 = *(const v4f*)(p + 16), a3 = *(const v4f*)(p + 20);
  Frag16 f;
#pragma unroll
  for (int i = 0; i < 4; ++i) {
    f.h[i]      = (_Float16)a0[i];
    f.h[4 + i]  = (_Float16)a1[i];
    f.h[8 + i]  = (_Float16)a2[i];
    f.h[12 + i] = (_Float16)a3[i];
  }
  return f.v;
}

// A-fragment from f16 memory: halves 0..7 <- p[0..7], halves 8..15 <- p[16..23]
DEV v16h load_a_f16(const _Float16* p) {
  Frag16 f;
  f.h8[0] = *(const v8h*)(p);
  f.h8[1] = *(const v8h*)(p + 16);
  return f.v;
}

// B-fragment from f16 memory: halves 0..7 <- p[0..7], halves 8..15 <- p[16..23]
DEV v16h load_b_f16(const _Float16* p) {
  Frag16 f;
  f.h8[0] = *(const v8h*)(p);
  f.h8[1] = *(const v8h*)(p + 16);
  return f.v;
}

// Async DMA one b128 (16 bytes) from global to LDS; tracked by ASYNCcnt.
// Generic LDS pointers carry the LDS byte offset in their low 32 bits.
DEV void async_copy_b128(const _Float16* gsrc, _Float16* ldst) {
  unsigned lds_addr = (unsigned)(size_t)ldst;
  unsigned long long gaddr = (unsigned long long)(size_t)gsrc;
  asm volatile("global_load_async_to_lds_b128 %0, %1, off"
               :: "v"(lds_addr), "v"(gaddr) : "memory");
}

DEV void wait_asynccnt0() {
  asm volatile("s_wait_asynccnt 0x0" ::: "memory");
}

// ---------------------------------------------------------------------------
// Kernel 0: one-shot weight conversion fp32 -> f16 (both weight matrices).
// ---------------------------------------------------------------------------
__global__ __launch_bounds__(256) void cvt_w_kernel(
    const float* __restrict__ wq, const float* __restrict__ wo,
    _Float16* __restrict__ wq16, _Float16* __restrict__ wo16) {
  const int idx = (blockIdx.x * 256 + threadIdx.x) * 8;
  const float* src;
  _Float16* dst;
  if (idx < N3_ * C_) { src = wq + idx;              dst = wq16 + idx; }
  else                { src = wo + (idx - N3_ * C_); dst = wo16 + (idx - N3_ * C_); }
  v4f a = *(const v4f*)(src), b = *(const v4f*)(src + 4);
  v8h o;
#pragma unroll
  for (int i = 0; i < 4; ++i) {
    o[i]     = (_Float16)a[i];
    o[4 + i] = (_Float16)b[i];
  }
  *(v8h*)dst = o;
}

// ---------------------------------------------------------------------------
// Kernel 1: fused QKV projection.  qkv = x @ W^T + b, split/reshaped to
// q,k,v in [B,H,T,Dh] f16 layout (q pre-scaled by log2(e)/sqrt(Dh)).
// Wave computes 16x64: one x A-fragment feeds 4 WMMAs vs f16 weights.
// ---------------------------------------------------------------------------
__global__ __launch_bounds__(128) void qkv_proj_kernel(
    const float* __restrict__ x, const _Float16* __restrict__ w16,
    const float* __restrict__ bias,
    _Float16* __restrict__ qf, _Float16* __restrict__ kf,
    _Float16* __restrict__ vf) {
  const int lane   = threadIdx.x & 31;
  const int wave   = threadIdx.x >> 5;
  const int m0     = (blockIdx.x * 4 + wave) * 16;
  const int n0     = blockIdx.y * 64;
  const int col    = lane & 15;
  const int hi     = lane >> 4;
  const int ka     = hi * 8;
  const int kb     = hi * 8;
  const int rowoff = hi * 8;

  v8f acc[4] = {};
  const float*    xrow  = x + (size_t)(m0 + col) * C_;
  const _Float16* wbase = w16 + (size_t)(n0 + col) * C_;
#pragma unroll
  for (int kk = 0; kk < C_; kk += 32) {
    v16h A = cvt_a_f32(xrow + kk + ka);
#pragma unroll
    for (int n = 0; n < 4; ++n)
      acc[n] = wmma_f16(A, load_b_f16(wbase + (size_t)n * 16 * C_ + kk + kb),
                        acc[n]);
  }

  const int batch = m0 / T_;
  const int t0    = (m0 % T_) + rowoff;
#pragma unroll
  for (int n = 0; n < 4; ++n) {
    const int o  = n0 + n * 16 + col;   // output feature in [0, 384)
    const float bb = bias[o];
    _Float16* dst;
    float scl = 1.0f;
    if (o < C_)          { dst = qf; scl = QSCALE_; }
    else if (o < 2 * C_) { dst = kf; }
    else                 { dst = vf; }
    const int head = (o >> 5) & 3;
    const int dh   = o & 31;
    size_t base = (((size_t)batch * H_ + head) * T_ + t0) * DH_ + dh;
#pragma unroll
    for (int i = 0; i < 8; ++i)
      dst[base + (size_t)i * DH_] = (_Float16)((acc[n][i] + bb) * scl);
  }
}

// ---------------------------------------------------------------------------
// Kernel 2: causal flash attention, transposed formulation.
//   S^T = K_tile * Q^T  -> queries in LANES, keys in REGISTERS
//   O^T = V^T * P^T     -> d in registers, queries in lanes
// K tiles: double-buffered LDS filled by GLOBAL_LOAD_ASYNC_TO_LDS_B128
// (ASYNCcnt-tracked DMA, overlapped with compute).  V tiles: register
// pipeline + transposed LDS stores.  P^T staged per-wave with b128 DS ops.
// Block = 4 waves = 64 query rows; 32 keys per step.
// ---------------------------------------------------------------------------
__global__ __launch_bounds__(128) void flash_attn_kernel(
    const _Float16* __restrict__ qf, const _Float16* __restrict__ kf,
    const _Float16* __restrict__ vf, _Float16* __restrict__ of) {
  __shared__ __attribute__((aligned(16))) _Float16 kTile[2][32 * 32];  // [key][d]
  __shared__ __attribute__((aligned(16))) _Float16 vTT[32 * 32];       // [d][key]
  __shared__ __attribute__((aligned(16))) _Float16 pT[4 * 16 * 32];    // [q][key]/wave

  const int tid    = threadIdx.x;
  const int lane   = tid & 31;
  const int wave   = tid >> 5;
  const int b      = blockIdx.z;
  const int h      = blockIdx.y;
  const int q0blk  = blockIdx.x * 64;
  const int q0     = q0blk + wave * 16;
  const int col    = lane & 15;    // query within wave tile (C/D col, B col)
  const int hi     = lane >> 4;
  const int ka     = hi * 8;       // A-fragment k-group offset
  const int kb     = hi * 8;       // B-fragment k-group offset
  const int rowoff = hi * 8;       // C/D row offset

  const size_t hb = ((size_t)b * H_ + h) * T_ * DH_;
  const _Float16* qbp = qf + hb;
  const _Float16* kbp = kf + hb;
  const _Float16* vbp = vf + hb;

  // B-fragment of Q^T: B[d][q] = Q[q][d]; d = kb..kb+7 and 16+kb..16+kb+7 per lane.
  const v16h Bq = load_b_f16(qbp + (size_t)(q0 + col) * DH_ + kb);

  v8f acc0 = {}, acc1 = {};          // O^T: d 0..15 tile and d 16..31 tile
  float mrow = -1e30f, lrow = 0.0f;  // per-lane (per-query) softmax state
  const int qidx = q0 + col;         // this lane's query index

  const int jend = q0blk + 64;       // block's causal extent (uniform)
  const int off  = tid * 8;          // cooperative-fill offset (16B per thread)
  const int fkey = off >> 5;         // key row this thread fills
  const int fd0  = off & 31;         // d-offset this thread fills

  // kick off async DMA of first K tile; preload first V tile into registers
  async_copy_b128(kbp + off, &kTile[0][off]);
  v8h vNxt = *(const v8h*)(vbp + off);
  int pbuf = 0;

  for (int j0 = 0; j0 < jend; j0 += 32) {
    wait_asynccnt0();                // this wave's K-tile DMA done
    __syncthreads();                 // all waves' DMA done; prev reads done
#pragma unroll
    for (int e = 0; e < 8; ++e)      // V tile transposed [d][key]
      vTT[(fd0 + e) * 32 + fkey] = vNxt[e];
    __syncthreads();                 // vTT + kTile[pbuf] ready

    if (j0 + 32 < jend) {            // pipeline next tile during compute
      async_copy_b128(kbp + (size_t)(j0 + 32) * DH_ + off,
                      &kTile[pbuf ^ 1][off]);
      vNxt = *(const v8h*)(vbp + (size_t)(j0 + 32) * DH_ + off);
    }

    const _Float16* kt = kTile[pbuf];
    // S^T = K * Q^T : two 16(key)x16(query) fp32 tiles
    v8f S0 = {}, S1 = {};
    S0 = wmma_f16(load_a_f16(&kt[(0  + col) * 32 + ka]), Bq, S0);
    S1 = wmma_f16(load_a_f16(&kt[(16 + col) * 32 + ka]), Bq, S1);

    // causal mask (key <= query) + running max, keys in registers
    float sv0[8], sv1[8];
    float mx = -1e30f;
#pragma unroll
    for (int i = 0; i < 8; ++i) {
      sv0[i] = (j0 + rowoff + i      <= qidx) ? S0[i] : -1e30f;
      sv1[i] = (j0 + 16 + rowoff + i <= qidx) ? S1[i] : -1e30f;
      mx = fmaxf(mx, fmaxf(sv0[i], sv1[i]));
    }
    mx = fmaxf(mx, __shfl_xor(mx, 16));   // combine complementary key half
    const float mnew  = fmaxf(mrow, mx);
    const float alpha = exp2f(mrow - mnew);

    // probabilities (log2 domain), packed to f16 for the P^T B-fragment
    v8h pA, pB;
    float rs = 0.0f;
#pragma unroll
    for (int i = 0; i < 8; ++i) {
      float p0 = exp2f(sv0[i] - mnew);
      float p1 = exp2f(sv1[i] - mnew);
      rs += p0 + p1;
      pA[i] = (_Float16)p0;
      pB[i] = (_Float16)p1;
    }
    rs += __shfl_xor(rs, 16);
    lrow = lrow * alpha + rs;
    mrow = mnew;
    acc0 *= alpha;
    acc1 *= alpha;

    // stage P^T [q][key] per wave: keys contiguous -> two b128 stores
    const int pbase = wave * 512 + col * 32;
    *(v8h*)&pT[pbase + rowoff]      = pA;
    *(v8h*)&pT[pbase + 16 + rowoff] = pB;

    // O^T += V^T * P^T (same-wave DS ops are in-order: RAW through pT safe)
    const v16h Bp = load_b_f16(&pT[wave * 512 + col * 32 + kb]);
    acc0 = wmma_f16(load_a_f16(&vTT[(0  + col) * 32 + ka]), Bp, acc0);
    acc1 = wmma_f16(load_a_f16(&vTT[(16 + col) * 32 + ka]), Bp, acc1);

    pbuf ^= 1;
  }

  // normalize (per-lane scalar) and write O: d contiguous -> b128 stores
  const float inv = 1.0f / lrow;
  v8h o0, o1;
#pragma unroll
  for (int i = 0; i < 8; ++i) {
    o0[i] = (_Float16)(acc0[i] * inv);
    o1[i] = (_Float16)(acc1[i] * inv);
  }
  _Float16* op = of + ((size_t)b * T_ + qidx) * C_ + h * DH_;
  *(v8h*)(op + rowoff)      = o0;   // d = rowoff..rowoff+7
  *(v8h*)(op + 16 + rowoff) = o1;   // d = 16+rowoff..
}

// ---------------------------------------------------------------------------
// Kernel 3: output projection.  out = O @ Wout^T + b, fp32 result.
// Wave computes 16x64; all operands f16 -> pure b128-load + WMMA inner loop.
// ---------------------------------------------------------------------------
__global__ __launch_bounds__(128) void out_proj_kernel(
    const _Float16* __restrict__ of, const _Float16* __restrict__ w16,
    const float* __restrict__ bias, float* __restrict__ out) {
  const int lane   = threadIdx.x & 31;
  const int wave   = threadIdx.x >> 5;
  const int m0     = (blockIdx.x * 4 + wave) * 16;
  const int n0     = blockIdx.y * 64;
  const int col    = lane & 15;
  const int hi     = lane >> 4;
  const int ka     = hi * 8;
  const int kb     = hi * 8;
  const int rowoff = hi * 8;

  v8f acc[4] = {};
  const _Float16* arow  = of + (size_t)(m0 + col) * C_;
  const _Float16* wbase = w16 + (size_t)(n0 + col) * C_;
#pragma unroll
  for (int kk = 0; kk < C_; kk += 32) {
    v16h A = load_a_f16(arow + kk + ka);
#pragma unroll
    for (int n = 0; n < 4; ++n)
      acc[n] = wmma_f16(A, load_b_f16(wbase + (size_t)n * 16 * C_ + kk + kb),
                        acc[n]);
  }

#pragma unroll
  for (int n = 0; n < 4; ++n) {
    const float bb = bias[n0 + n * 16 + col];
#pragma unroll
    for (int i = 0; i < 8; ++i)
      out[(size_t)(m0 + rowoff + i) * C_ + n0 + n * 16 + col] = acc[n][i] + bb;
  }
}

// ---------------------------------------------------------------------------
extern "C" void kernel_launch(void* const* d_in, const int* in_sizes, int n_in,
                              void* d_out, int out_size, void* d_ws, size_t ws_size,
                              hipStream_t stream) {
  (void)in_sizes; (void)n_in; (void)out_size; (void)ws_size;

  const float* x     = (const float*)d_in[0];
  const float* qkv_w = (const float*)d_in[1];
  const float* qkv_b = (const float*)d_in[2];
  const float* out_w = (const float*)d_in[3];
  const float* out_b = (const float*)d_in[4];
  float* out = (float*)d_out;

  _Float16* ws = (_Float16*)d_ws;
  const size_t SEG = (size_t)B_ * H_ * T_ * DH_;  // 1,048,576 halves (2 MB)
  _Float16* qf   = ws;
  _Float16* kf   = ws + SEG;
  _Float16* vf   = ws + 2 * SEG;
  _Float16* of   = ws + 3 * SEG;
  _Float16* wq16 = ws + 4 * SEG;                  // 49152 halves
  _Float16* wo16 = wq16 + (size_t)N3_ * C_;       // 16384 halves

  cvt_w_kernel<<<dim3((N3_ * C_ + C_ * C_) / 2048), 256, 0, stream>>>(
      qkv_w, out_w, wq16, wo16);
  qkv_proj_kernel<<<dim3(M_ / 64, N3_ / 64), 128, 0, stream>>>(
      x, wq16, qkv_b, qf, kf, vf);
  flash_attn_kernel<<<dim3(T_ / 64, H_, B_), 128, 0, stream>>>(
      qf, kf, vf, of);
  out_proj_kernel<<<dim3(M_ / 64, C_ / 64), 128, 0, stream>>>(
      of, wo16, out_b, out);
}
